// MultiTaskNet_34419867910604
// MI455X (gfx1250) — matrix operand order repaired and checked by reading, not compiled, not run on hardware
//
#include <hip/hip_runtime.h>
#include <hip/hip_bf16.h>
#include <stdint.h>

// ---- CDNA5 WMMA types ------------------------------------------------------
typedef __attribute__((ext_vector_type(16))) __bf16 bf16x16;
typedef __attribute__((ext_vector_type(8)))  float  f32x8;

#define N_CLS 20
#define HW    3136        // 56*56
#define F6    4096
#define NPR   200         // real proto rows (4 global + 196 local)
#define MT    13          // 208 padded rows / 16
#define KT    128         // 4096 / 32
#define PIXT  196         // 3136 / 16 pixel tiles
#define TEMP_ 20.0f

static __device__ __forceinline__ unsigned short f2bf(float f) {
  unsigned u = __float_as_uint(f);
  return (unsigned short)((u + 0x7FFFu + ((u >> 16) & 1u)) >> 16);  // RNE
}

// ---- spatial mean over H*W, one block per (b,f) plane ----------------------
__global__ __launch_bounds__(256) void mean_hw_kernel(const float* __restrict__ in,
                                                      float* __restrict__ out, int hw) {
  __shared__ float red[256];
  int plane = blockIdx.x;
  const float* p = in + (size_t)plane * hw;
  float acc = 0.f;
  for (int i = threadIdx.x; i < hw; i += 256) acc += p[i];
  red[threadIdx.x] = acc;
  __syncthreads();
  for (int s = 128; s > 0; s >>= 1) {
    if (threadIdx.x < s) red[threadIdx.x] += red[threadIdx.x + s];
    __syncthreads();
  }
  if (threadIdx.x == 0) out[plane] = red[0] / (float)hw;
}

// ---- mask prep: class map, per-(c,b) area, valid-cell bits, first batch ----
__global__ __launch_bounds__(256) void mask_prep_kernel(const int* __restrict__ pm,
                                                        int* __restrict__ clsmap,
                                                        float* __restrict__ area,
                                                        int* __restrict__ firstb,
                                                        unsigned* __restrict__ lvalid) {
  __shared__ int      sArea[N_CLS * 4];
  __shared__ unsigned sVB[N_CLS * 4 * 7];
  int tid = threadIdx.x;
  for (int i = tid; i < N_CLS * 4; i += 256) sArea[i] = 0;
  for (int i = tid; i < N_CLS * 4 * 7; i += 256) sVB[i] = 0u;
  __syncthreads();
  for (int w = tid; w < 4 * 196; w += 256) {
    int b = w / 196, cell = w % 196;
    int gy = cell / 14, gx = cell % 14;
    int cl[16];
#pragma unroll
    for (int dy = 0; dy < 4; ++dy)
#pragma unroll
      for (int dx = 0; dx < 4; ++dx) {
        int iy = gy * 4 + dy, ix = gx * 4 + dx;
        int c = pm[(size_t)b * 448 * 448 + (size_t)(iy * 8) * 448 + ix * 8];
        cl[dy * 4 + dx] = c;
        clsmap[b * HW + iy * 56 + ix] = c;
      }
#pragma unroll
    for (int c = 0; c < N_CLS; ++c) {
      int n = 0;
#pragma unroll
      for (int j = 0; j < 16; ++j) n += (cl[j] == c);
      if (n) atomicAdd(&sArea[c * 4 + b], n);
      if (n > 8) atomicOr(&sVB[(c * 4 + b) * 7 + (cell >> 5)], 1u << (cell & 31));
    }
  }
  __syncthreads();
  for (int i = tid; i < N_CLS * 4; i += 256) area[i] = (float)sArea[i];
  for (int c = tid; c < N_CLS; c += 256) {
    int fb = 0;
    for (int b = 3; b >= 0; --b) {
      unsigned any = 0;
      for (int i = 0; i < 7; ++i) any |= sVB[(c * 4 + b) * 7 + i];
      if (any) fb = b;                       // keeps the FIRST b with any valid cell
    }
    firstb[c] = fb;
    for (int i = 0; i < 7; ++i) lvalid[c * 7 + i] = sVB[(c * 4 + fb) * 7 + i];
  }
}

// ---- one fused pass over conv6: gproto, cand (4x4 pooled), mean6 -----------
__global__ __launch_bounds__(256) void proto_accum_kernel(const float* __restrict__ conv6,
                                                          const int* __restrict__ clsmap,
                                                          const float* __restrict__ area,
                                                          float* __restrict__ gproto,
                                                          float* __restrict__ cand,
                                                          float* __restrict__ mean6) {
  __shared__ float sCls[21];
  __shared__ float sCell[196];
  int tid = threadIdx.x;
  int f = blockIdx.x;          // 0..4095
  int b = blockIdx.y;          // 0..3
  if (tid < 21) sCls[tid] = 0.f;
  if (tid >= 32 && tid < 32 + 196) sCell[tid - 32] = 0.f;
  __syncthreads();
  const float* src = conv6 + ((size_t)b * F6 + f) * HW;
  const int*   cm  = clsmap + b * HW;
  for (int pix = tid; pix < HW; pix += 256) {
    float v = src[pix];
    int cls = cm[pix];                               // 0..20 (20 = "no class")
    atomicAdd(&sCls[cls], v);                        // ds_add_f32
    int iy = pix / 56, ix = pix - iy * 56;
    atomicAdd(&sCell[(iy >> 2) * 14 + (ix >> 2)], v);
  }
  __syncthreads();
  if (tid < N_CLS)
    gproto[((size_t)tid * 4 + b) * F6 + f] = sCls[tid] / (area[tid * 4 + b] + 1e-6f);
  if (tid >= 32 && tid < 32 + 196)
    cand[((size_t)b * F6 + f) * 196 + (tid - 32)] = sCell[tid - 32] * (1.f / 16.f);
  if (tid == 255) {
    float t = 0.f;
    for (int c = 0; c < 21; ++c) t += sCls[c];
    mean6[b * F6 + f] = t * (1.f / (float)HW);
  }
}

// ---- tiny FC heads: x1, x2, x3 (4x20 each) ---------------------------------
__global__ __launch_bounds__(256) void fc_heads_kernel(const float* __restrict__ m45,
                                                       const float* __restrict__ m52,
                                                       const float* __restrict__ m6,
                                                       const float* __restrict__ w1,
                                                       const float* __restrict__ bi1,
                                                       const float* __restrict__ w2,
                                                       const float* __restrict__ bi2,
                                                       const float* __restrict__ w8,
                                                       float* __restrict__ out) {
  int t = threadIdx.x;
  if (t < 80) {
    int b = t / 20, c = t % 20;
    float s = bi1[c];
    for (int f = 0; f < 512; ++f) s = fmaf(m45[b * 512 + f], w1[c * 512 + f], s);
    out[t] = s;
  } else if (t < 160) {
    int u = t - 80, b = u / 20, c = u % 20;
    float s = bi2[c];
    for (int f = 0; f < 1024; ++f) s = fmaf(m52[b * 1024 + f], w2[c * 1024 + f], s);
    out[80 + u] = s;
  } else if (t < 240) {
    int u = t - 160, b = u / 20, c = u % 20;
    float s = 0.f;
    for (int f = 0; f < F6; ++f) s = fmaf(m6[b * F6 + f], w8[c * F6 + f], s);
    out[160 + u] = s;
  }
}

// ---- cam = relu(w_fc8 . conv6), lane = pixel, weights chunked in LDS -------
__global__ __launch_bounds__(256) void cam_kernel(const float* __restrict__ conv6,
                                                  const float* __restrict__ w8,
                                                  float* __restrict__ cam) {
  __shared__ float sW[N_CLS * 128];
  int b = blockIdx.y;
  int pix0 = blockIdx.x * 256 + threadIdx.x;
  int pix = min(pix0, HW - 1);
  float acc[N_CLS];
#pragma unroll
  for (int c = 0; c < N_CLS; ++c) acc[c] = 0.f;
  for (int f0 = 0; f0 < F6; f0 += 128) {
    __syncthreads();
    for (int i = threadIdx.x; i < N_CLS * 128; i += 256)
      sW[i] = w8[(size_t)(i >> 7) * F6 + f0 + (i & 127)];
    __syncthreads();
    for (int fo = 0; fo < 128; ++fo) {
      float v = conv6[((size_t)b * F6 + f0 + fo) * HW + pix];
#pragma unroll
      for (int c = 0; c < N_CLS; ++c) acc[c] = fmaf(v, sW[c * 128 + fo], acc[c]);
    }
  }
  if (pix0 < HW) {
#pragma unroll
    for (int c = 0; c < N_CLS; ++c)
      cam[((size_t)b * N_CLS + c) * HW + pix0] = fmaxf(acc[c], 0.f);
  }
}

// ---- per-pixel 1/||q||_2 ---------------------------------------------------
__global__ __launch_bounds__(256) void qnorm_kernel(const float* __restrict__ q,
                                                    float* __restrict__ qinv) {
  int b = blockIdx.y;
  int pix = blockIdx.x * 256 + threadIdx.x;
  if (pix >= HW) return;
  const float* p = q + (size_t)b * F6 * HW + pix;
  float acc = 0.f;
  for (int f = 0; f < F6; ++f) { float v = p[(size_t)f * HW]; acc = fmaf(v, v, acc); }
  qinv[b * HW + pix] = 1.f / fmaxf(sqrtf(acc), 1e-12f);
}

// ---- pack qn into bf16 B-fragments (ISA 7.12.2: 32x16 16-bit B layout) -----
// lane L: column N = L&15, K = (e>=8?16:0) + (L>=16?8:0) + (e&7)  (e = 0..15)
__global__ __launch_bounds__(256) void pack_qn_kernel(const float* __restrict__ q,
                                                      const float* __restrict__ qinv,
                                                      unsigned short* __restrict__ qnB) {
  int wave = threadIdx.x >> 5, lane = threadIdx.x & 31;
  int gid = blockIdx.x * 8 + wave;              // b*PIXT*KT waves
  int kt = gid & 127;
  int t  = (gid >> 7) % PIXT;
  int b  = gid / (PIXT * KT);
  int half = lane >> 4, n = lane & 15;
  int pix = t * 16 + n;
  float inv = qinv[b * HW + pix];
  alignas(16) unsigned short o[16];
#pragma unroll
  for (int e = 0; e < 16; ++e) {
    int f = kt * 32 + ((e >= 8) ? 16 : 0) + half * 8 + (e & 7);
    o[e] = f2bf(q[((size_t)b * F6 + f) * HW + pix] * inv);
  }
  uint4* dst = (uint4*)(qnB + (((size_t)(b * PIXT + t) * KT + kt) * 512 + lane * 16));
  dst[0] = *(const uint4*)&o[0];
  dst[1] = *(const uint4*)&o[8];
}

// ---- per-proto-row 1/||p||_2 -----------------------------------------------
__global__ __launch_bounds__(256) void pnorm_kernel(const float* __restrict__ gproto,
                                                    const float* __restrict__ cand,
                                                    const int* __restrict__ firstb,
                                                    float* __restrict__ pinv) {
  __shared__ float red[256];
  int p = blockIdx.x, c = blockIdx.y;          // p < 200
  int fb = firstb[c];
  float acc = 0.f;
  for (int f = threadIdx.x; f < F6; f += 256) {
    float v = (p < 4) ? gproto[((size_t)c * 4 + p) * F6 + f]
                      : cand[((size_t)fb * F6 + f) * 196 + (p - 4)];
    acc = fmaf(v, v, acc);
  }
  red[threadIdx.x] = acc;
  __syncthreads();
  for (int s = 128; s > 0; s >>= 1) {
    if (threadIdx.x < s) red[threadIdx.x] += red[threadIdx.x + s];
    __syncthreads();
  }
  if (threadIdx.x == 0) pinv[c * NPR + p] = 1.f / fmaxf(sqrtf(red[0]), 1e-12f);
}

// ---- pack pn into bf16 A-fragments (ISA 7.12.2: 16x32 16-bit A layout) -----
// lane L: row M = L&15, K = (e>=8?16:0) + (L>=16?8:0) + (e&7)
__global__ __launch_bounds__(256) void pack_pn_kernel(const float* __restrict__ gproto,
                                                      const float* __restrict__ cand,
                                                      const int* __restrict__ firstb,
                                                      const float* __restrict__ pinv,
                                                      unsigned short* __restrict__ pnA) {
  int wave = threadIdx.x >> 5, lane = threadIdx.x & 31;
  int gid = blockIdx.x * 8 + wave;              // c*KT*MT waves
  int mt = gid % MT;
  int kt = (gid / MT) & 127;
  int c  = gid / (MT * KT);
  int half = lane >> 4, mrow = lane & 15;
  int p = mt * 16 + mrow;
  int fb = firstb[c];
  float inv = (p < NPR) ? pinv[c * NPR + p] : 0.f;
  alignas(16) unsigned short o[16];
#pragma unroll
  for (int e = 0; e < 16; ++e) {
    int f = kt * 32 + ((e >= 8) ? 16 : 0) + half * 8 + (e & 7);
    float v = 0.f;
    if (p < 4)        v = gproto[((size_t)c * 4 + p) * F6 + f] * inv;
    else if (p < NPR) v = cand[((size_t)fb * F6 + f) * 196 + (p - 4)] * inv;
    o[e] = f2bf(v);
  }
  // fragment-major, mt innermost so the 13 frags of one (c,kt) are contiguous
  uint4* dst = (uint4*)(pnA + (((size_t)(c * KT + kt) * MT + mt) * 512 + lane * 16));
  dst[0] = *(const uint4*)&o[0];
  dst[1] = *(const uint4*)&o[8];
}

// ---- main fused kernel: sims GEMM (bf16 WMMA) + masked softmax + reduce ----
// Double-buffered LDS A-chunk (one barrier per K-step); A fragments and the
// B fragment are software-pipelined so every v_wmma has its operands in
// flight one step ahead (no dscnt==0 stall per WMMA).
__global__ __launch_bounds__(256) void sims_softmax_kernel(const unsigned short* __restrict__ pnA,
                                                           const unsigned short* __restrict__ qnB,
                                                           const unsigned* __restrict__ lvalid,
                                                           float* __restrict__ seg) {
  __shared__ alignas(16) unsigned short ldsA[2][MT * 512];   // 2 x 13 KB
  int wave = threadIdx.x >> 5, lane = threadIdx.x & 31;
  int grp = blockIdx.x;                 // 0..24 pixel groups (8 tiles each)
  int c   = blockIdx.y;                 // 0..19
  int b   = blockIdx.z;                 // 0..3
  int tileIdx = grp * 8 + wave;         // 0..199 (196..199 are pad work)
  int tcl = min(tileIdx, PIXT - 1);

  union AB { uint4 u[2]; bf16x16 v; };

  f32x8 acc[MT];
#pragma unroll
  for (int m = 0; m < MT; ++m)
#pragma unroll
    for (int r = 0; r < 8; ++r) acc[m][r] = 0.f;

  const size_t qbase = ((size_t)(b * PIXT + tcl) * KT) * 512 + lane * 16;

  // prologue: stage A(kt=0) into LDS buf0, load B(kt=0) into regs
  {
    const uint4* sA = (const uint4*)(pnA + ((size_t)(c * KT) * MT) * 512);
    if (threadIdx.x < 208) {
      uint4* d = ((uint4*)ldsA[0]) + threadIdx.x * 4;
      const uint4* s = sA + threadIdx.x * 4;
      d[0] = s[0]; d[1] = s[1]; d[2] = s[2]; d[3] = s[3];
    }
  }
  AB Bb[2];
  {
    const uint4* bq = (const uint4*)(qnB + qbase);
    Bb[0].u[0] = bq[0]; Bb[0].u[1] = bq[1];
  }
  __syncthreads();

#pragma unroll 2
  for (int kt = 0; kt < KT; ++kt) {
    int cur = kt & 1, nxt = cur ^ 1;

    // issue next-step global loads EARLY; park in registers through the WMMAs
    uint4 st0, st1, st2, st3;
    bool staging = (kt + 1 < KT) && (threadIdx.x < 208);
    if (kt + 1 < KT) {
      const uint4* bq = (const uint4*)(qnB + qbase + (size_t)(kt + 1) * 512);
      Bb[nxt].u[0] = bq[0]; Bb[nxt].u[1] = bq[1];
    }
    if (staging) {
      const uint4* s = (const uint4*)(pnA + ((size_t)(c * KT + kt + 1) * MT) * 512)
                       + threadIdx.x * 4;
      st0 = s[0]; st1 = s[1]; st2 = s[2]; st3 = s[3];
    }
    if (kt + 2 < KT) {
      const char* nx = (const char*)(pnA + ((size_t)(c * KT + kt + 2) * MT) * 512);
      __builtin_prefetch(nx + (size_t)threadIdx.x * 52, 0, 1);   // global_prefetch_b8
    }

    // compute: double-buffered A fragments from LDS
    const uint4* lb = (const uint4*)ldsA[cur];
    AB Af[2];
    Af[0].u[0] = lb[lane * 2]; Af[0].u[1] = lb[lane * 2 + 1];
#pragma unroll
    for (int m = 0; m < MT; ++m) {
      if (m + 1 < MT) {
        const uint4* ap = lb + (m + 1) * 64 + lane * 2;
        Af[(m + 1) & 1].u[0] = ap[0]; Af[(m + 1) & 1].u[1] = ap[1];
      }
      acc[m] = __builtin_amdgcn_wmma_f32_16x16x32_bf16(
          /*neg_a=*/false, Af[m & 1].v, /*neg_b=*/false, Bb[cur].v,
          /*c_mod=*/(short)0, acc[m], /*reuse_a=*/false, /*reuse_b=*/false);
    }

    // drain staged data into the other LDS buffer, then the single barrier
    if (staging) {
      uint4* d = ((uint4*)ldsA[nxt]) + threadIdx.x * 4;
      d[0] = st0; d[1] = st1; d[2] = st2; d[3] = st3;
    }
    __syncthreads();
  }

  // ---- fused masked softmax over the proto dimension (per pixel column) ----
  unsigned vb[7];
#pragma unroll
  for (int i = 0; i < 7; ++i) vb[i] = lvalid[c * 7 + i];
  int moff = (lane >= 16) ? 8 : 0;      // C layout: lane L holds M = r + (L>=16?8:0)

  float mx = -3.0e38f;
#pragma unroll
  for (int m = 0; m < MT; ++m)
#pragma unroll
    for (int r = 0; r < 8; ++r) {
      int p = m * 16 + r + moff;
      bool val = (p < 4) || (p < NPR && ((vb[(p - 4) >> 5] >> ((p - 4) & 31)) & 1u));
      float s = acc[m][r] * TEMP_;
      if (val) mx = fmaxf(mx, s);
    }
  mx = fmaxf(mx, __shfl_xor(mx, 16, 32));   // combine the two M-halves

  float se = 0.f, ss = 0.f;
#pragma unroll
  for (int m = 0; m < MT; ++m)
#pragma unroll
    for (int r = 0; r < 8; ++r) {
      int p = m * 16 + r + moff;
      bool val = (p < 4) || (p < NPR && ((vb[(p - 4) >> 5] >> ((p - 4) & 31)) & 1u));
      if (val) {
        float s = acc[m][r] * TEMP_;
        float e = __expf(s - mx);
        se += e; ss = fmaf(e, s, ss);
      }
    }
  se += __shfl_xor(se, 16, 32);
  ss += __shfl_xor(ss, 16, 32);

  if (lane < 16 && tileIdx < PIXT) {
    int pix = tileIdx * 16 + lane;      // N = lane (0..15)
    seg[((size_t)b * N_CLS + c) * HW + pix] = ss / se;
  }
}

// ---------------------------------------------------------------------------
extern "C" void kernel_launch(void* const* d_in, const int* in_sizes, int n_in,
                              void* d_out, int out_size, void* d_ws, size_t ws_size,
                              hipStream_t stream) {
  const float* b45   = (const float*)d_in[0];
  const float* b52   = (const float*)d_in[1];
  const float* conv6 = (const float*)d_in[2];
  const float* qf    = (const float*)d_in[3];
  const float* w1    = (const float*)d_in[4];
  const float* bi1   = (const float*)d_in[5];
  const float* w2    = (const float*)d_in[6];
  const float* bi2   = (const float*)d_in[7];
  const float* w8    = (const float*)d_in[8];
  const int*   pm    = (const int*)d_in[9];
  float* out = (float*)d_out;

  char* ws = (char*)d_ws;
  size_t off = 0;
  auto wsalloc = [&](size_t bytes) -> void* {
    void* p = ws + off;
    off = (off + bytes + 255) & ~(size_t)255;
    return p;
  };

  float*          m45    = (float*)wsalloc((size_t)4 * 512 * 4);
  float*          m52    = (float*)wsalloc((size_t)4 * 1024 * 4);
  float*          m6     = (float*)wsalloc((size_t)4 * F6 * 4);
  int*            clsmap = (int*)wsalloc((size_t)4 * HW * 4);
  float*          area   = (float*)wsalloc((size_t)N_CLS * 4 * 4);
  int*            firstb = (int*)wsalloc((size_t)N_CLS * 4);
  unsigned*       lvalid = (unsigned*)wsalloc((size_t)N_CLS * 7 * 4);
  float*          gproto = (float*)wsalloc((size_t)N_CLS * 4 * F6 * 4);
  float*          cand   = (float*)wsalloc((size_t)4 * F6 * 196 * 4);
  float*          pinv   = (float*)wsalloc((size_t)N_CLS * NPR * 4);
  float*          qinv   = (float*)wsalloc((size_t)4 * HW * 4);
  unsigned short* pnA    = (unsigned short*)wsalloc((size_t)N_CLS * KT * MT * 512 * 2);
  unsigned short* qnB    = (unsigned short*)wsalloc((size_t)4 * PIXT * KT * 512 * 2);

  mean_hw_kernel<<<4 * 512, 256, 0, stream>>>(b45, m45, 112 * 112);
  mean_hw_kernel<<<4 * 1024, 256, 0, stream>>>(b52, m52, 56 * 56);
  mask_prep_kernel<<<1, 256, 0, stream>>>(pm, clsmap, area, firstb, lvalid);
  proto_accum_kernel<<<dim3(F6, 4), 256, 0, stream>>>(conv6, clsmap, area, gproto, cand, m6);
  fc_heads_kernel<<<1, 256, 0, stream>>>(m45, m52, m6, w1, bi1, w2, bi2, w8, out);
  cam_kernel<<<dim3(13, 4), 256, 0, stream>>>(conv6, w8, out + 240);
  qnorm_kernel<<<dim3(13, 4), 256, 0, stream>>>(qf, qinv);
  pack_qn_kernel<<<(4 * PIXT * KT) / 8, 256, 0, stream>>>(qf, qinv, qnB);
  pnorm_kernel<<<dim3(NPR, N_CLS), 256, 0, stream>>>(gproto, cand, firstb, pinv);
  pack_pn_kernel<<<(N_CLS * KT * MT) / 8, 256, 0, stream>>>(gproto, cand, firstb, pinv, pnA);
  sims_softmax_kernel<<<dim3(25, N_CLS, 4), 256, 0, stream>>>(
      pnA, qnB, lvalid, out + 240 + (size_t)4 * N_CLS * HW);
}
